// TUPEMultiHeadAttention_gem_13881334300860
// MI455X (gfx1250) — matrix operand order repaired and checked by reading, not compiled, not run on hardware
//
#include <hip/hip_runtime.h>
#include <hip/hip_bf16.h>

// ---------------------------------------------------------------------------
// TUPE multi-head attention for gfx1250 (MI455X), f16 WMMA w/ f32 accumulate.
//   B=4, S=1024, D=1024, H=16, DEPTH=64
// Pipeline:
//   (1) f32->f16 conversion of activations / weights / pos table
//   (2) WMMA GEMMs (64x64 block tile, 4 WMMA/wave/K-step): q_c,k_c,v_c ; q_p,k_p
//   (3) fused flash attention (content + pos scores + theta overrides,
//       streaming softmax, P@V) -> O (f16)
//   (4) WMMA GEMM: out = O@Wo + bo (f32 -> d_out)
// Workspace layout (74 MB total, MB-aligned regions):
//   qh 0 | kh 8M | vh 16M | Wqh 24M | Wkh 26M | Wvh 28M | Uqh 30M | Ukh 32M
//   Woh 34M | posh 36M | qch 38M | kch 46M | vch 54M | qph 62M | kph 64M | oh 66M
// ---------------------------------------------------------------------------

#define S_LEN   1024
#define D_MODEL 1024
#define NHEADS  16
#define DEPTH   64
#define BATCH   4

typedef _Float16 v16h __attribute__((ext_vector_type(16)));
typedef float    v8f  __attribute__((ext_vector_type(8)));

union FragH {
    v16h v;
    unsigned u[8];
    _Float16 h[16];
};

static __device__ __forceinline__ v8f wmma_f16(const FragH& a, const FragH& b, v8f c) {
    // D = A(16x32 f16) x B(32x16 f16) + C(16x16 f32)
    return __builtin_amdgcn_wmma_f32_16x16x32_f16(
        false, a.v, false, b.v, (short)0, c, false, false);
}

static __device__ __forceinline__ v8f zero8() {
    v8f r;
#pragma unroll
    for (int i = 0; i < 8; ++i) r[i] = 0.0f;
    return r;
}

static __device__ __forceinline__ float rmax16(float x) {
#pragma unroll
    for (int m = 1; m < 16; m <<= 1) x = fmaxf(x, __shfl_xor(x, m));
    return x;  // masks 1,2,4,8 never cross the 16-lane half of the wave32
}
static __device__ __forceinline__ float rsum16(float x) {
#pragma unroll
    for (int m = 1; m < 16; m <<= 1) x += __shfl_xor(x, m);
    return x;
}

// ---------------------------------------------------------------------------
// f32 -> f16 conversion (grid-stride)
// ---------------------------------------------------------------------------
__global__ void cvt_f32_f16(const float* __restrict__ src, _Float16* __restrict__ dst, int n) {
    int i = blockIdx.x * blockDim.x + threadIdx.x;
    int stride = gridDim.x * blockDim.x;
    for (; i < n; i += stride) dst[i] = (_Float16)src[i];
}

// ---------------------------------------------------------------------------
// C[M,N] = A[M,K](f16) @ W[K,N](f16) + bias.
// Block = 4 waves (128 thr), block tile 64(M) x 64(N); wave wv owns rows
// 16*wv..16*wv+15 and all 64 columns -> 4 WMMA per K-step (K step = 32).
// ---------------------------------------------------------------------------
__global__ __launch_bounds__(128) void gemm_f16_wmma(
    const _Float16* __restrict__ A, const _Float16* __restrict__ W,
    const float* __restrict__ bias, _Float16* __restrict__ outh,
    float* __restrict__ outf, int M, int N, int K) {

    __shared__ _Float16 As[64][34];   // A tile rows m0..m0+63, cols kk..kk+31 (pad 34)
    __shared__ _Float16 Bst[64][34];  // B tile transposed: Bst[n][k], n0..n0+63

    const int tid  = threadIdx.x;
    const int lane = tid & 31;
    const int wv   = tid >> 5;
    const int hi   = lane >> 4;   // 0 for lanes 0-15, 1 for lanes 16-31
    const int nI   = lane & 15;
    const int m0   = blockIdx.y * 64;
    const int n0   = blockIdx.x * 64;

    v8f acc[4];
#pragma unroll
    for (int t = 0; t < 4; ++t) acc[t] = zero8();

    for (int kk = 0; kk < K; kk += 32) {
        // prefetch next K tile while this one is staged (global_prefetch_b8)
        if (kk + 32 < K) {
            __builtin_prefetch(A + (size_t)(m0 + (tid & 63)) * K + kk + 32 + (tid >> 6) * 16, 0, 0);
            __builtin_prefetch(W + (size_t)(kk + 32 + (tid & 31)) * N + n0 + (tid >> 5) * 16, 0, 0);
        }
        __syncthreads();
        // stage A: 64x32 f16 = 1024 u32; 8 u32 per thread
#pragma unroll
        for (int i = 0; i < 8; ++i) {
            int u   = tid * 8 + i;
            int row = u >> 4;           // 16 u32 per 32-f16 row
            int cp  = u & 15;
            *(unsigned*)&As[row][2 * cp] =
                *(const unsigned*)(A + (size_t)(m0 + row) * K + kk + 2 * cp);
        }
        // stage B transposed: 32(k) x 64(n) f16; 8 n-pairs per thread
#pragma unroll
        for (int i = 0; i < 8; ++i) {
            int e  = tid * 8 + i;       // 0..1023
            int k  = e >> 5;            // 32 pairs per k-row
            int n  = (e & 31) * 2;
            const _Float16* wp = W + (size_t)(kk + k) * N + n0 + n;
            Bst[n][k]     = wp[0];
            Bst[n + 1][k] = wp[1];
        }
        __syncthreads();

        FragH a;
#pragma unroll
        for (int v = 0; v < 8; ++v) {
            const int KA = 2 * v + 16 * hi;                       // A frag K pair
            a.u[v] = *(const unsigned*)&As[16 * wv + nI][KA];
        }
#pragma unroll
        for (int t = 0; t < 4; ++t) {
            FragH b;
#pragma unroll
            for (int v = 0; v < 8; ++v) {
                const int KB = 2 * v + 16 * hi;                   // B frag K pair
                b.u[v] = *(const unsigned*)&Bst[16 * t + nI][KB];
            }
            acc[t] = wmma_f16(a, b, acc[t]);
        }
    }

#pragma unroll
    for (int t = 0; t < 4; ++t) {
        const int col  = n0 + 16 * t + nI;
        const float bc = bias ? bias[col] : 0.0f;
#pragma unroll
        for (int r = 0; r < 8; ++r) {
            const int row = m0 + 16 * wv + r + 8 * hi;  // C layout: M = r + 8*(lane/16)
            const float val = acc[t][r] + bc;
            if (outh) outh[(size_t)row * N + col] = (_Float16)val;
            if (outf) outf[(size_t)row * N + col] = val;
        }
    }
}

// ---------------------------------------------------------------------------
// Fused flash attention. One wave per block; block = (q-tile of 16 rows, head, batch).
// Per iteration: 32 keys -> 8 score WMMAs (content+pos over K=64) + 4 PV WMMAs.
// ---------------------------------------------------------------------------
__global__ __launch_bounds__(32) void tupe_flash_attn(
    const _Float16* __restrict__ qc, const _Float16* __restrict__ kc,
    const _Float16* __restrict__ vc, const _Float16* __restrict__ qp,
    const _Float16* __restrict__ kp, const float* __restrict__ thcc,
    const float* __restrict__ thco, const float* __restrict__ thoc,
    _Float16* __restrict__ oh) {

    __shared__ _Float16 p16[16][36];  // P tile (16 q-rows x 32 keys), padded
    __shared__ _Float16 vt[64][36];   // V tile transposed: vt[depth][key]

    const int lane = threadIdx.x & 31;
    const int hi   = lane >> 4;
    const int nI   = lane & 15;
    const int q0   = blockIdx.x * 16;
    const int h    = blockIdx.y;
    const int bz   = blockIdx.z;

    const float th_cc = thcc[h];
    const float th_co = thco[h];
    const float th_oc = thoc[h];

    // Q fragments (A 16x64 -> two K=32 frags), content + positional
    FragH qa0, qa1, pa0, pa1;
    {
        const _Float16* qrow = qc + ((size_t)(bz * S_LEN + q0 + nI)) * D_MODEL + h * DEPTH;
        const _Float16* prow = qp + ((size_t)(q0 + nI)) * D_MODEL + h * DEPTH;
#pragma unroll
        for (int v = 0; v < 8; ++v) {
            const int K0 = 2 * v + 16 * hi;
            qa0.u[v] = *(const unsigned*)(qrow + K0);
            qa1.u[v] = *(const unsigned*)(qrow + 32 + K0);
            pa0.u[v] = *(const unsigned*)(prow + K0);
            pa1.u[v] = *(const unsigned*)(prow + 32 + K0);
        }
    }

    v8f accd[4];
#pragma unroll
    for (int t = 0; t < 4; ++t) accd[t] = zero8();
    float mrow[8], lrow[8];
#pragma unroll
    for (int r = 0; r < 8; ++r) { mrow[r] = -3.0e38f; lrow[r] = 0.0f; }

    for (int j = 0; j < S_LEN; j += 32) {
        // prefetch next iteration's K/V rows (global_prefetch_b8)
        if (j + 32 < S_LEN) {
            __builtin_prefetch(
                kc + ((size_t)(bz * S_LEN + j + 32 + lane)) * D_MODEL + h * DEPTH, 0, 0);
            __builtin_prefetch(
                vc + ((size_t)(bz * S_LEN + j + 32 + lane)) * D_MODEL + h * DEPTH, 0, 0);
        }

        // ---- scores: two 16x16 tiles (t = 0,1), K = DEPTH = 64 ----
        v8f stile[2];
#pragma unroll
        for (int t = 0; t < 2; ++t) {
            FragH kb0, kb1, pb0, pb1;
            const int key = j + 16 * t + nI;  // B frag: column N = lane%16
            const _Float16* kcr = kc + ((size_t)(bz * S_LEN + key)) * D_MODEL + h * DEPTH;
            const _Float16* kpr = kp + ((size_t)key) * D_MODEL + h * DEPTH;
#pragma unroll
            for (int v = 0; v < 8; ++v) {
                const int K0 = 2 * v + 16 * hi;  // B frag: K = 2v + 16*(lane/16) (+h)
                kb0.u[v] = *(const unsigned*)(kcr + K0);
                kb1.u[v] = *(const unsigned*)(kcr + 32 + K0);
                pb0.u[v] = *(const unsigned*)(kpr + K0);
                pb1.u[v] = *(const unsigned*)(kpr + 32 + K0);
            }
            v8f c = zero8();
            c = wmma_f16(qa0, kb0, c);
            c = wmma_f16(qa1, kb1, c);
            v8f d = zero8();
            d = wmma_f16(pa0, pb0, d);
            d = wmma_f16(pa1, pb1, d);
            // theta overrides on positional part, then combine + scale
#pragma unroll
            for (int r = 0; r < 8; ++r) {
                const int qpos = q0 + r + 8 * hi;
                const int kpos = j + 16 * t + nI;
                float pos = d[r];
                if (kpos == 0)      pos = (qpos == 0) ? th_cc : th_oc;
                else if (qpos == 0) pos = th_co;
                stile[t][r] = (c[r] + pos) * 0.125f;  // 1/sqrt(64)
            }
        }

        // ---- streaming softmax: row stats via half-wave shuffles ----
#pragma unroll
        for (int r = 0; r < 8; ++r) {
            float mx = rmax16(fmaxf(stile[0][r], stile[1][r]));
            const float mnew = fmaxf(mrow[r], mx);
            const float p0 = __expf(stile[0][r] - mnew);
            const float p1 = __expf(stile[1][r] - mnew);
            const float corr = __expf(mrow[r] - mnew);
            lrow[r] = lrow[r] * corr + rsum16(p0 + p1);
            mrow[r] = mnew;
#pragma unroll
            for (int t = 0; t < 4; ++t) accd[t][r] *= corr;
            // C-layout -> LDS row-major (row = r + 8*hi, col = key-in-block)
            p16[r + 8 * hi][nI]      = (_Float16)p0;
            p16[r + 8 * hi][16 + nI] = (_Float16)p1;
        }

        // ---- stage V tile transposed: lane owns key row (j+lane) ----
        {
            const _Float16* vrow =
                vc + ((size_t)(bz * S_LEN + j + lane)) * D_MODEL + h * DEPTH;
#pragma unroll
            for (int dd = 0; dd < DEPTH; ++dd) vt[dd][lane] = vrow[dd];
        }
        __builtin_amdgcn_wave_barrier();  // keep LDS store->load program order

        // ---- P(16x32) @ V(32x64): A frag from p16, B frags from vt ----
        FragH pA;
#pragma unroll
        for (int v = 0; v < 8; ++v) {
            const int K0 = 2 * v + 16 * hi;
            pA.u[v] = *(const unsigned*)&p16[nI][K0];
        }
#pragma unroll
        for (int t = 0; t < 4; ++t) {
            FragH vB;
#pragma unroll
            for (int v = 0; v < 8; ++v) {
                const int K0 = 2 * v + 16 * hi;
                vB.u[v] = *(const unsigned*)&vt[t * 16 + nI][K0];
            }
            accd[t] = wmma_f16(pA, vB, accd[t]);
        }
        __builtin_amdgcn_wave_barrier();
    }

    // ---- normalize and store O (f16) ----
#pragma unroll
    for (int t = 0; t < 4; ++t) {
#pragma unroll
        for (int r = 0; r < 8; ++r) {
            const float o = accd[t][r] / lrow[r];
            oh[((size_t)(bz * S_LEN + q0 + r + 8 * hi)) * D_MODEL +
               h * DEPTH + t * 16 + nI] = (_Float16)o;
        }
    }
}

// ---------------------------------------------------------------------------
extern "C" void kernel_launch(void* const* d_in, const int* in_sizes, int n_in,
                              void* d_out, int out_size, void* d_ws, size_t ws_size,
                              hipStream_t stream) {
    const float* q   = (const float*)d_in[0];
    const float* k   = (const float*)d_in[1];
    const float* v   = (const float*)d_in[2];
    const float* Wq  = (const float*)d_in[3];
    const float* bq  = (const float*)d_in[4];
    const float* Wk  = (const float*)d_in[5];
    const float* bk  = (const float*)d_in[6];
    const float* Wv  = (const float*)d_in[7];
    const float* bv  = (const float*)d_in[8];
    const float* Uq  = (const float*)d_in[9];
    const float* buq = (const float*)d_in[10];
    const float* Uk  = (const float*)d_in[11];
    const float* buk = (const float*)d_in[12];
    const float* pos = (const float*)d_in[13];
    const float* thcc = (const float*)d_in[14];
    const float* thco = (const float*)d_in[15];
    const float* thoc = (const float*)d_in[16];
    const float* Wo  = (const float*)d_in[17];
    const float* bo  = (const float*)d_in[18];

    const size_t MB = 1u << 20;
    char* ws = (char*)d_ws;
    _Float16* qh   = (_Float16*)(ws + 0 * MB);
    _Float16* kh   = (_Float16*)(ws + 8 * MB);
    _Float16* vh   = (_Float16*)(ws + 16 * MB);
    _Float16* Wqh  = (_Float16*)(ws + 24 * MB);
    _Float16* Wkh  = (_Float16*)(ws + 26 * MB);
    _Float16* Wvh  = (_Float16*)(ws + 28 * MB);
    _Float16* Uqh  = (_Float16*)(ws + 30 * MB);
    _Float16* Ukh  = (_Float16*)(ws + 32 * MB);
    _Float16* Woh  = (_Float16*)(ws + 34 * MB);
    _Float16* posh = (_Float16*)(ws + 36 * MB);
    _Float16* qch  = (_Float16*)(ws + 38 * MB);
    _Float16* kch  = (_Float16*)(ws + 46 * MB);
    _Float16* vch  = (_Float16*)(ws + 54 * MB);
    _Float16* qph  = (_Float16*)(ws + 62 * MB);
    _Float16* kph  = (_Float16*)(ws + 64 * MB);
    _Float16* oh   = (_Float16*)(ws + 66 * MB);

    const int nBS = BATCH * S_LEN * D_MODEL;   // 4,194,304
    const int nW  = D_MODEL * D_MODEL;         // 1,048,576
    const int nP  = S_LEN * D_MODEL;           // 1,048,576 (first S rows of pos_table)

    dim3 cb(256), cg(2048);
    cvt_f32_f16<<<cg, cb, 0, stream>>>(q,   qh,   nBS);
    cvt_f32_f16<<<cg, cb, 0, stream>>>(k,   kh,   nBS);
    cvt_f32_f16<<<cg, cb, 0, stream>>>(v,   vh,   nBS);
    cvt_f32_f16<<<cg, cb, 0, stream>>>(Wq,  Wqh,  nW);
    cvt_f32_f16<<<cg, cb, 0, stream>>>(Wk,  Wkh,  nW);
    cvt_f32_f16<<<cg, cb, 0, stream>>>(Wv,  Wvh,  nW);
    cvt_f32_f16<<<cg, cb, 0, stream>>>(Uq,  Uqh,  nW);
    cvt_f32_f16<<<cg, cb, 0, stream>>>(Uk,  Ukh,  nW);
    cvt_f32_f16<<<cg, cb, 0, stream>>>(Wo,  Woh,  nW);
    cvt_f32_f16<<<cg, cb, 0, stream>>>(pos, posh, nP);

    const int M_act = BATCH * S_LEN;  // 4096
    dim3 gb(128);
    dim3 gActs(D_MODEL / 64, M_act / 64);   // (16, 64)
    dim3 gPos (D_MODEL / 64, S_LEN / 64);   // (16, 16)

    gemm_f16_wmma<<<gActs, gb, 0, stream>>>(qh,   Wqh, bq,  qch, nullptr, M_act, D_MODEL, D_MODEL);
    gemm_f16_wmma<<<gActs, gb, 0, stream>>>(kh,   Wkh, bk,  kch, nullptr, M_act, D_MODEL, D_MODEL);
    gemm_f16_wmma<<<gActs, gb, 0, stream>>>(vh,   Wvh, bv,  vch, nullptr, M_act, D_MODEL, D_MODEL);
    gemm_f16_wmma<<<gPos,  gb, 0, stream>>>(posh, Uqh, buq, qph, nullptr, S_LEN, D_MODEL, D_MODEL);
    gemm_f16_wmma<<<gPos,  gb, 0, stream>>>(posh, Ukh, buk, kph, nullptr, S_LEN, D_MODEL, D_MODEL);

    dim3 ab(32), ag(S_LEN / 16, NHEADS, BATCH);  // (64, 16, 4)
    tupe_flash_attn<<<ag, ab, 0, stream>>>(qch, kch, vch, qph, kph, thcc, thco, thoc, oh);

    gemm_f16_wmma<<<gActs, gb, 0, stream>>>(oh, Woh, bo, nullptr, (float*)d_out,
                                            M_act, D_MODEL, D_MODEL);
}
